// AddAttention_25701084299721
// MI455X (gfx1250) — matrix operand order repaired and checked by reading, not compiled, not run on hardware
//
#include <hip/hip_runtime.h>
#include <hip/hip_bf16.h>
#include <math.h>

typedef __attribute__((ext_vector_type(16))) _Float16 v16h;
typedef __attribute__((ext_vector_type(8)))  _Float16 v8h;
typedef __attribute__((ext_vector_type(8)))  float    v8f;

#define KSEL 30
#define KNRM 8
#define CDIM 128
#define KNN_THREADS 128

// ---------------------------------------------------------------------------
// Kernel 0: one-time f32 -> f16 conversion of the Q/K weight matrices so the
// attention kernel loads B-fragments as packed f16 (b128 loads, no v_cvt in
// the WMMA loop). Weights are broadcast to 8192 waves and stay L2-resident.
// ---------------------------------------------------------------------------
__global__ __launch_bounds__(256)
void cvt_weights_kernel(const float* __restrict__ wq_w,
                        const float* __restrict__ wk_w,
                        _Float16* __restrict__ wq16,
                        _Float16* __restrict__ wk16, int n)
{
    const int i = blockIdx.x * blockDim.x + threadIdx.x;
    if (i < n) {
        wq16[i] = (_Float16)wq_w[i];
        wk16[i] = (_Float16)wk_w[i];
    }
}

// ---------------------------------------------------------------------------
// Kernel 1: per-point top-30 nearest voxels (squared distance, ascending).
// One block (128 threads) per query point. Register insertion sort per
// thread, then a 30-round LDS argmin tournament merge. Indices are emitted
// nearest-first, matching lax.top_k(-dist) ordering (idx8 = first 8).
// ---------------------------------------------------------------------------
__global__ __launch_bounds__(KNN_THREADS)
void knn_topk_kernel(const float* __restrict__ x_world,
                     const float* __restrict__ voxel_point,
                     int* __restrict__ out_idx, int N, int M)
{
    const int n   = blockIdx.x;
    const int tid = threadIdx.x;
    if (n >= N) return;

    const float px = x_world[n * 3 + 0];
    const float py = x_world[n * 3 + 1];
    const float pz = x_world[n * 3 + 2];

    float bd[KSEL];
    int   bi[KSEL];
#pragma unroll
    for (int k = 0; k < KSEL; ++k) { bd[k] = 3.4e38f; bi[k] = 0; }

    for (int m = tid; m < M; m += KNN_THREADS) {
        const float dx = px - voxel_point[m * 3 + 0];
        const float dy = py - voxel_point[m * 3 + 1];
        const float dz = pz - voxel_point[m * 3 + 2];
        const float d = dx * dx + dy * dy + dz * dz;
        if (d < bd[KSEL - 1]) {
            float cd = d; int ci = m;
#pragma unroll
            for (int k = 0; k < KSEL; ++k) {
                const bool sw = cd < bd[k];
                const float tf = bd[k];
                const int   ti = bi[k];
                if (sw) { bd[k] = cd; cd = tf; bi[k] = ci; ci = ti; }
            }
        }
    }

    __shared__ float cdist[KNN_THREADS * KSEL];
    __shared__ int   cidx [KNN_THREADS * KSEL];
    __shared__ int   ptrs [KNN_THREADS];
    __shared__ float rd   [KNN_THREADS];
    __shared__ int   ri   [KNN_THREADS];

#pragma unroll
    for (int k = 0; k < KSEL; ++k) {
        cdist[tid * KSEL + k] = bd[k];
        cidx [tid * KSEL + k] = bi[k];
    }
    ptrs[tid] = 0;
    __syncthreads();

    for (int r = 0; r < KSEL; ++r) {
        const int p = ptrs[tid];
        rd[tid] = (p < KSEL) ? cdist[tid * KSEL + p] : 3.4e38f;
        ri[tid] = tid;
        __syncthreads();
#pragma unroll
        for (int s = KNN_THREADS / 2; s > 0; s >>= 1) {
            if (tid < s) {
                if (rd[tid + s] < rd[tid]) { rd[tid] = rd[tid + s]; ri[tid] = ri[tid + s]; }
            }
            __syncthreads();
        }
        if (tid == 0) {
            const int w  = ri[0];
            const int pw = ptrs[w];
            out_idx[n * KSEL + r] = cidx[w * KSEL + pw];
            ptrs[w] = pw + 1;
        }
        __syncthreads();
    }
}

// Assemble a v16h A/B fragment from two contiguous 8-half (16B) runs.
__device__ __forceinline__ v16h cat8(v8h lo, v8h hi)
{
    return __builtin_shufflevector(lo, hi, 0, 1, 2, 3, 4, 5, 6, 7,
                                           8, 9, 10, 11, 12, 13, 14, 15);
}

// ---------------------------------------------------------------------------
// Kernel 2: gather + fc + Q/K + attention + mean, one wave32 per point.
// All 32 lanes live -> EXEC all-ones around every v_wmma.
// Fragment accesses are 16-byte vectors:
//   A (16x32 f16):  lane = {row = lane&15, kHalf = lane>>4};
//                   elements = h[row][kbase .. kbase+7] ++ h[row][kbase+8 .. +15]
//                   with kbase = ks*32 + kHalf*16 (two b128 LDS loads).
//   B (32x16 f16):  elements = W[col][ks*32 + kHalf*16 .. +15]
//                   (one contiguous 32B run -> two b128 loads).
//   C/D f32:        row = mt*16 + kHalf*8 + r, col = lane&15.
// ---------------------------------------------------------------------------
__global__ __launch_bounds__(32)
void attn_kernel(const float* __restrict__ x_world,
                 const float* __restrict__ voxel_point,
                 const float* __restrict__ voxel_normal,
                 const float* __restrict__ v,
                 const float* __restrict__ fc_w, const float* __restrict__ fc_b,
                 const _Float16* __restrict__ wq16, const float* __restrict__ wq_b,
                 const _Float16* __restrict__ wk16, const float* __restrict__ wk_b,
                 const int* __restrict__ knn_idx,
                 float* __restrict__ out, int N)
{
    const int n    = blockIdx.x;
    const int lane = threadIdx.x;
    if (n >= N) return;

    __shared__ _Float16 hbuf[32][CDIM];   // fc output  (rows 30,31 zero)
    __shared__ _Float16 qbuf[32][CDIM];   // Q
    __shared__ _Float16 kbuf[32][CDIM];   // K
    __shared__ float    sbuf[32][32];     // scores
    __shared__ float    vn_sh[32][3];
    __shared__ float    xf_sh[32][6];
    __shared__ float    vsel_sh[32];
    __shared__ float    fcw_sh[CDIM * 6];
    __shared__ float    osum[32];

    // Warm L2 for the broadcast f16 weights (global_prefetch_b8).
    __builtin_prefetch(wq16 + lane * 512, 0, 1);
    __builtin_prefetch(wk16 + lane * 512, 0, 1);

    for (int i = lane; i < CDIM * 6; i += 32) fcw_sh[i] = fc_w[i];

    const float px = x_world[n * 3 + 0];
    const float py = x_world[n * 3 + 1];
    const float pz = x_world[n * 3 + 2];

    int   myidx = 0;
    float vpx = 0.f, vpy = 0.f, vpz = 0.f, vnx = 0.f, vny = 0.f, vnz = 0.f, vv = 0.f;
    if (lane < KSEL) {
        myidx = knn_idx[n * KSEL + lane];
        vpx = voxel_point[myidx * 3 + 0];
        vpy = voxel_point[myidx * 3 + 1];
        vpz = voxel_point[myidx * 3 + 2];
        vnx = voxel_normal[myidx * 3 + 0];
        vny = voxel_normal[myidx * 3 + 1];
        vnz = voxel_normal[myidx * 3 + 2];
        vv  = v[myidx];
    }
    vn_sh[lane][0] = vnx; vn_sh[lane][1] = vny; vn_sh[lane][2] = vnz;
    vsel_sh[lane]  = (lane < KSEL) ? vv : 0.f;
    __syncthreads();

    // x_normal = mean of the 8 nearest voxel normals.
    float nx = 0.f, ny = 0.f, nz = 0.f;
#pragma unroll
    for (int j = 0; j < KNRM; ++j) {
        nx += vn_sh[j][0]; ny += vn_sh[j][1]; nz += vn_sh[j][2];
    }
    const float inv8 = 1.0f / (float)KNRM;
    nx *= inv8; ny *= inv8; nz *= inv8;

    const bool valid = (lane < KSEL);
    xf_sh[lane][0] = valid ? (px - vpx) : 0.f;
    xf_sh[lane][1] = valid ? (py - vpy) : 0.f;
    xf_sh[lane][2] = valid ? (pz - vpz) : 0.f;
    xf_sh[lane][3] = valid ? (nx - vnx) : 0.f;
    xf_sh[lane][4] = valid ? (ny - vny) : 0.f;
    xf_sh[lane][5] = valid ? (nz - vnz) : 0.f;
    __syncthreads();

    // fc: row = lane, 128 outputs, K=6 (too small for WMMA -> VALU).
    {
        const float x0 = xf_sh[lane][0], x1 = xf_sh[lane][1], x2 = xf_sh[lane][2];
        const float x3 = xf_sh[lane][3], x4 = xf_sh[lane][4], x5 = xf_sh[lane][5];
        for (int c = 0; c < CDIM; ++c) {
            float acc = fc_b[c];
            acc += x0 * fcw_sh[c * 6 + 0] + x1 * fcw_sh[c * 6 + 1]
                 + x2 * fcw_sh[c * 6 + 2] + x3 * fcw_sh[c * 6 + 3]
                 + x4 * fcw_sh[c * 6 + 4] + x5 * fcw_sh[c * 6 + 5];
            hbuf[lane][c] = (_Float16)(valid ? acc : 0.f);
        }
    }
    __syncthreads();

    const int rowA  = lane & 15;
    const int kHalf = lane >> 4;   // 0 or 1

    // Q = h @ Wq^T + bq ; K = h @ Wk^T + bk   (32x128 @ 128x128)
    for (int which = 0; which < 2; ++which) {
        const _Float16* W = which ? wk16 : wq16;
        const float*    B = which ? wk_b : wq_b;
        _Float16 (*obuf)[CDIM] = which ? kbuf : qbuf;
#pragma unroll
        for (int mt = 0; mt < 2; ++mt) {
            v16h afr[4];
#pragma unroll
            for (int ks = 0; ks < 4; ++ks) {
                const int kbase = ks * 32 + kHalf * 16;
                const v8h lo = *(const v8h*)&hbuf[mt * 16 + rowA][kbase];
                const v8h hi = *(const v8h*)&hbuf[mt * 16 + rowA][kbase + 8];
                afr[ks] = cat8(lo, hi);
            }
            for (int nt = 0; nt < 8; ++nt) {
                const int col = nt * 16 + (lane & 15);
                const float bb = B[col];
                v8f acc;
#pragma unroll
                for (int r = 0; r < 8; ++r) acc[r] = bb;
#pragma unroll
                for (int ks = 0; ks < 4; ++ks) {
                    const v8h* wp = (const v8h*)(W + col * CDIM + ks * 32 + kHalf * 16);
                    const v16h bfr = cat8(wp[0], wp[1]);
                    acc = __builtin_amdgcn_wmma_f32_16x16x32_f16(
                        false, afr[ks], false, bfr, (short)0, acc, false, false);
                }
#pragma unroll
                for (int r = 0; r < 8; ++r) {
                    const int row = mt * 16 + kHalf * 8 + r;
                    obuf[row][col] = (_Float16)acc[r];
                }
            }
        }
        __syncthreads();
    }

    // scores = Q @ K^T / sqrt(128)   (32x128 @ 128x32)
    const float scale = 0.08838834764831845f;
#pragma unroll
    for (int mt = 0; mt < 2; ++mt) {
        v16h afr[4];
#pragma unroll
        for (int ks = 0; ks < 4; ++ks) {
            const int kbase = ks * 32 + kHalf * 16;
            const v8h lo = *(const v8h*)&qbuf[mt * 16 + rowA][kbase];
            const v8h hi = *(const v8h*)&qbuf[mt * 16 + rowA][kbase + 8];
            afr[ks] = cat8(lo, hi);
        }
#pragma unroll
        for (int nt = 0; nt < 2; ++nt) {
            v8f acc = {};
#pragma unroll
            for (int ks = 0; ks < 4; ++ks) {
                // B[k][n] = K[n][k] -> contiguous run of kbuf[col][...]
                const v8h* kp = (const v8h*)&kbuf[nt * 16 + (lane & 15)][ks * 32 + kHalf * 16];
                const v16h bfr = cat8(kp[0], kp[1]);
                acc = __builtin_amdgcn_wmma_f32_16x16x32_f16(
                    false, afr[ks], false, bfr, (short)0, acc, false, false);
            }
#pragma unroll
            for (int r = 0; r < 8; ++r) {
                sbuf[mt * 16 + kHalf * 8 + r][nt * 16 + (lane & 15)] = acc[r] * scale;
            }
        }
    }
    __syncthreads();

    // Row softmax over 30 keys + attn @ v, then mean over 30 rows.
    float o = 0.f;
    if (lane < KSEL) {
        float mmax = -3.4e38f;
        for (int j = 0; j < KSEL; ++j) mmax = fmaxf(mmax, sbuf[lane][j]);
        float s = 0.f, acc = 0.f;
        for (int j = 0; j < KSEL; ++j) {
            const float e = __expf(sbuf[lane][j] - mmax);
            s   += e;
            acc += e * vsel_sh[j];
        }
        o = acc / s;
    }
    osum[lane] = (lane < KSEL) ? o : 0.f;
    __syncthreads();
    if (lane == 0) {
        float t = 0.f;
        for (int j = 0; j < KSEL; ++j) t += osum[j];
        out[n] = t / (float)KSEL;
    }
}

// ---------------------------------------------------------------------------
extern "C" void kernel_launch(void* const* d_in, const int* in_sizes, int n_in,
                              void* d_out, int out_size, void* d_ws, size_t ws_size,
                              hipStream_t stream)
{
    const float* x_world      = (const float*)d_in[0];
    const float* voxel_point  = (const float*)d_in[1];
    const float* voxel_normal = (const float*)d_in[2];
    const float* v            = (const float*)d_in[3];
    const float* fc_w         = (const float*)d_in[4];
    const float* fc_b         = (const float*)d_in[5];
    const float* wq_w         = (const float*)d_in[6];
    const float* wq_b         = (const float*)d_in[7];
    const float* wk_w         = (const float*)d_in[8];
    const float* wk_b         = (const float*)d_in[9];

    const int N = in_sizes[0] / 3;
    const int M = in_sizes[1] / 3;

    // Workspace layout: [ knn_idx: N*KSEL int ][ wq16: 128*128 f16 ][ wk16 ]
    char* ws = (char*)d_ws;
    int*      knn_idx = (int*)ws;
    size_t    off     = (size_t)N * KSEL * sizeof(int);
    off = (off + 255) & ~(size_t)255;
    _Float16* wq16 = (_Float16*)(ws + off);
    _Float16* wk16 = (_Float16*)(ws + off + CDIM * CDIM * sizeof(_Float16));

    cvt_weights_kernel<<<(CDIM * CDIM + 255) / 256, 256, 0, stream>>>(
        wq_w, wk_w, wq16, wk16, CDIM * CDIM);
    knn_topk_kernel<<<N, KNN_THREADS, 0, stream>>>(x_world, voxel_point, knn_idx, N, M);
    attn_kernel<<<N, 32, 0, stream>>>(x_world, voxel_point, voxel_normal, v,
                                      fc_w, fc_b, wq16, wq_b, wk16, wk_b,
                                      knn_idx, (float*)d_out, N);
}
